// Fourier_57174604644855
// MI455X (gfx1250) — compile-verified, matrix operand order checked by reading, not run on hardware
//
#include <hip/hip_runtime.h>
#include <stdint.h>

#define SIG    1000
#define M_ROWS 8192
#define KP     1024   // padded K for GEMM1
#define NPSD   512    // padded psd width (cols 0..499 live, rest zero)
#define K2     512    // padded K for GEMM2
#define NOUT   141
#define NT2    9      // ceil(141/16)

typedef __attribute__((ext_vector_type(16))) __bf16 v16bf;
typedef __attribute__((ext_vector_type(8)))  float  v8f;

union Frag {
  uint32_t u[8];
  v16bf    v;
};

__device__ __forceinline__ uint16_t f2bf(float f) {
  uint32_t u = __float_as_uint(f);
  u += 0x7FFFu + ((u >> 16) & 1u);   // round-to-nearest-even
  return (uint16_t)(u >> 16);
}

// A[m][k] = bf16(x[m][k] * ham[k]), zero-padded to K=1024
__global__ void prep_A(const float* __restrict__ x, const float* __restrict__ filt,
                       uint16_t* __restrict__ A) {
  for (int64_t i = (int64_t)blockIdx.x * blockDim.x + threadIdx.x;
       i < (int64_t)M_ROWS * KP;
       i += (int64_t)gridDim.x * blockDim.x) {
    int k = (int)(i & (KP - 1));
    int m = (int)(i >> 10);
    float v = 0.0f;
    if (k < SIG) v = x[(int64_t)m * SIG + k] * filt[k];
    A[i] = f2bf(v);
  }
}

// BT[n][k]: n in [0,500) -> Wf[k][n] (cos bins), n in [512,1012) -> Wf[k][1000+(n-512)] (-sin bins)
__global__ void prep_BT(const float* __restrict__ Wf, uint16_t* __restrict__ BT) {
  for (int64_t i = (int64_t)blockIdx.x * blockDim.x + threadIdx.x;
       i < (int64_t)KP * KP;
       i += (int64_t)gridDim.x * blockDim.x) {
    int k = (int)(i & (KP - 1));
    int n = (int)(i >> 10);
    float v = 0.0f;
    if (k < SIG) {
      if (n < 500)                   v = Wf[(int64_t)k * 2000 + n];
      else if (n >= 512 && n < 1012) v = Wf[(int64_t)k * 2000 + 1000 + (n - 512)];
    }
    BT[i] = f2bf(v);
  }
}

// W3b[n][k] = bf16(W3[n][k]), padded to 144x512 (W3 row n == GEMM2 B column n)
__global__ void prep_W3(const float* __restrict__ W3, uint16_t* __restrict__ W3b) {
  for (int64_t i = (int64_t)blockIdx.x * blockDim.x + threadIdx.x;
       i < (int64_t)144 * K2;
       i += (int64_t)gridDim.x * blockDim.x) {
    int k = (int)(i & (K2 - 1));
    int n = (int)(i >> 9);
    float v = (n < NOUT && k < 500) ? W3[(int64_t)n * 500 + k] : 0.0f;
    W3b[i] = f2bf(v);
  }
}

// 16-bit fragment loader per documented layout:
// lanes 0-15 hold K 0..7 & 16..23, lanes 16-31 hold K 8..15 & 24..31 -> two b128 loads.
__device__ __forceinline__ Frag load_frag(const uint16_t* __restrict__ base,
                                          int elem_off) {
  const uint4* p = (const uint4*)(base + elem_off);     // elem_off % 8 == 0
  uint4 lo = p[0];
  uint4 hi = p[2];                                      // +16 bf16 elements
  Frag f;
  f.u[0] = lo.x; f.u[1] = lo.y; f.u[2] = lo.z; f.u[3] = lo.w;
  f.u[4] = hi.x; f.u[5] = hi.y; f.u[6] = hi.z; f.u[7] = hi.w;
  return f;
}

// GEMM1 fused with PSD. Register-blocked: each wave computes a 64(M) x 32(N) psd slab,
// i.e. 4 M-tiles x 2 N-tiles, each N-tile with an R (cols n) and I (cols n+512) accumulator.
// Per K-step: 4 A frags + 4 B frags -> 16 WMMAs (16 B/lane loaded per WMMA).
#define MT1 4
#define NT1 2
__global__ void __launch_bounds__(256) gemm1_psd(const uint16_t* __restrict__ A,
                                                 const uint16_t* __restrict__ BT,
                                                 uint16_t* __restrict__ PSD) {
  const int lane = threadIdx.x & 31;
  const int wave = threadIdx.x >> 5;
  const int gw   = blockIdx.x * 8 + wave;      // 2048 waves total
  const int mt   = gw >> 4;                    // 128 M-slabs (64 rows each)
  const int nt   = gw & 15;                    // 16 N-slabs (32 psd cols each)
  const int m0   = mt << 6;
  const int n0   = nt << 5;

  const int l15  = lane & 15;
  const int ksel = (lane >> 4) << 3;           // 0 for lanes 0-15, 8 for lanes 16-31

  const uint16_t* arow0 = A  + (int64_t)(m0 + l15) * KP;             // +16*KP per M-tile
  const uint16_t* brow0 = BT + (int64_t)(n0 + l15) * KP;             // R tiles
  const uint16_t* irow0 = BT + (int64_t)(n0 + 512 + l15) * KP;       // I tiles

  v8f accR[MT1][NT1];
  v8f accI[MT1][NT1];
#pragma unroll
  for (int mi = 0; mi < MT1; ++mi)
#pragma unroll
    for (int ni = 0; ni < NT1; ++ni) { accR[mi][ni] = {}; accI[mi][ni] = {}; }

  for (int kb = 0; kb < KP; kb += 32) {
    const int off = kb + ksel;
    Frag bR[NT1], bI[NT1];
#pragma unroll
    for (int ni = 0; ni < NT1; ++ni) {
      bR[ni] = load_frag(brow0 + (int64_t)(ni * 16) * KP, off);
      bI[ni] = load_frag(irow0 + (int64_t)(ni * 16) * KP, off);
    }
#pragma unroll
    for (int mi = 0; mi < MT1; ++mi) {
      Frag a = load_frag(arow0 + (int64_t)(mi * 16) * KP, off);
#pragma unroll
      for (int ni = 0; ni < NT1; ++ni) {
        accR[mi][ni] = __builtin_amdgcn_wmma_f32_16x16x32_bf16(
            false, a.v, false, bR[ni].v, (short)0, accR[mi][ni], false, false);
        accI[mi][ni] = __builtin_amdgcn_wmma_f32_16x16x32_bf16(
            false, a.v, false, bI[ni].v, (short)0, accI[mi][ni], false, false);
      }
    }
  }

  // C/D layout: VGPR i, lanes 0-15 -> row +i, lanes 16-31 -> row +8+i; col = lane%16
  const int rhalf = (lane >> 4) << 3;
#pragma unroll
  for (int mi = 0; mi < MT1; ++mi) {
#pragma unroll
    for (int ni = 0; ni < NT1; ++ni) {
      const int rbase = m0 + mi * 16 + rhalf;
      const int pcol  = n0 + ni * 16 + l15;
#pragma unroll
      for (int i = 0; i < 8; ++i) {
        float r  = accR[mi][ni][i];
        float im = accI[mi][ni][i];
        float p  = (r * r + im * im) * 1e-6f;  // / SIG^2
        PSD[(int64_t)(rbase + i) * NPSD + pcol] = f2bf(p);
      }
    }
  }
}

// GEMM2: out = psd @ W3^T + b3 (8192x512 bf16 x 512x144 -> 8192x141 f32).
// Register-blocked 64(M) x 16(N): 4 A frags + 1 B frag -> 4 WMMAs per K-step.
#define MT2 4
__global__ void __launch_bounds__(256) gemm2_out(const uint16_t* __restrict__ PSD,
                                                 const uint16_t* __restrict__ W3b,
                                                 const float* __restrict__ b3,
                                                 float* __restrict__ out) {
  const int lane = threadIdx.x & 31;
  const int wave = threadIdx.x >> 5;
  const int gw   = blockIdx.x * 8 + wave;      // 1152 waves total
  const int mt   = gw / NT2;                   // 128 M-slabs (64 rows each)
  const int nt   = gw % NT2;                   // 9 N-tiles
  const int m0   = mt << 6;
  const int n0   = nt << 4;

  const int l15  = lane & 15;
  const int ksel = (lane >> 4) << 3;

  const uint16_t* arow0 = PSD + (int64_t)(m0 + l15) * NPSD;
  const uint16_t* brow  = W3b + (int64_t)(n0 + l15) * K2;

  v8f acc[MT2];
#pragma unroll
  for (int mi = 0; mi < MT2; ++mi) acc[mi] = {};

  for (int kb = 0; kb < K2; kb += 32) {
    const int off = kb + ksel;
    Frag b = load_frag(brow, off);
#pragma unroll
    for (int mi = 0; mi < MT2; ++mi) {
      Frag a = load_frag(arow0 + (int64_t)(mi * 16) * NPSD, off);
      acc[mi] = __builtin_amdgcn_wmma_f32_16x16x32_bf16(
          false, a.v, false, b.v, (short)0, acc[mi], false, false);
    }
  }

  const int rhalf = (lane >> 4) << 3;
  const int col   = n0 + l15;
  if (col < NOUT) {
    float bias = b3[col];
#pragma unroll
    for (int mi = 0; mi < MT2; ++mi) {
      const int rbase = m0 + mi * 16 + rhalf;
#pragma unroll
      for (int i = 0; i < 8; ++i) {
        out[(int64_t)(rbase + i) * NOUT + col] = acc[mi][i] + bias;
      }
    }
  }
}

extern "C" void kernel_launch(void* const* d_in, const int* in_sizes, int n_in,
                              void* d_out, int out_size, void* d_ws, size_t ws_size,
                              hipStream_t stream) {
  const float* x    = (const float*)d_in[0];  // 8192*1000
  const float* filt = (const float*)d_in[1];  // 1000
  const float* Wf   = (const float*)d_in[2];  // 1000*2000
  // d_in[3] (Wfc) unused: i2 == -i1 analytically, pi == 0, psd == (r^2+i^2)/SIG^2
  const float* W3   = (const float*)d_in[4];  // 141*500
  const float* b3   = (const float*)d_in[5];  // 141

  uint8_t* ws = (uint8_t*)d_ws;
  uint16_t* A   = (uint16_t*)(ws);                         // 8192*1024*2 = 16 MiB
  uint16_t* BT  = (uint16_t*)(ws + (size_t)16777216);      // 1024*1024*2 =  2 MiB
  uint16_t* W3b = (uint16_t*)(ws + (size_t)18874368);      //  144*512*2  = 144 KiB
  uint16_t* PSD = (uint16_t*)(ws + (size_t)19021824);      // 8192*512*2  =  8 MiB
                                                           // total ~26.1 MiB

  prep_A <<<2048, 256, 0, stream>>>(x, filt, A);
  prep_BT<<<1024, 256, 0, stream>>>(Wf, BT);
  prep_W3<<<288,  256, 0, stream>>>(W3, W3b);

  gemm1_psd<<<256, 256, 0, stream>>>(A, BT, PSD);                   // 2048 waves
  gemm2_out<<<144, 256, 0, stream>>>(PSD, W3b, b3, (float*)d_out);  // 1152 waves
}
